// Block_85134841741625
// MI455X (gfx1250) — matrix operand order repaired and checked by reading, not compiled, not run on hardware
//
#include <hip/hip_runtime.h>
#include <hip/hip_bf16.h>
#include <cstdint>
#include <cstddef>

// Problem constants (match reference)
#define B_  4
#define T_  1024
#define C_  2048
#define H_  32
#define N_  64
#define F_  7168
#define BT_ (B_ * T_)
#define EPS_ 1e-5f

typedef __attribute__((ext_vector_type(16))) __bf16 bf16x16;
typedef __attribute__((ext_vector_type(8)))  float  f32x8;

__device__ __forceinline__ __bf16 f2bf(float f) {
  unsigned u = __builtin_bit_cast(unsigned, f);
  u = (u + 0x7FFFu + ((u >> 16) & 1u)) >> 16;   // round-to-nearest-even
  unsigned short s = (unsigned short)u;
  return __builtin_bit_cast(__bf16, s);
}

// CDNA5 async HBM->LDS copy (16B per lane), tracked by ASYNCcnt.
// lds_off: wave-relative LDS byte address (low 32 bits of the flat address).
__device__ __forceinline__ void async_ld16(uint32_t lds_off, const void* g) {
  asm volatile("global_load_async_to_lds_b128 %0, %1, off"
               :: "v"(lds_off), "v"(g) : "memory");
}
__device__ __forceinline__ void wait_async0() {
  asm volatile("s_wait_asynccnt 0x0" ::: "memory");
}

// ---------------------------------------------------------------------------
// Transpose + f32->bf16 convert:  W (K x N row-major)  ->  Wt (N x K row-major)
// ---------------------------------------------------------------------------
__global__ __launch_bounds__(256)
void k_transpose_bf16(const float* __restrict__ W, __bf16* __restrict__ Wt,
                      int K, int N) {
  __shared__ float tile[32][33];
  const int kb = blockIdx.y * 32, nb = blockIdx.x * 32;
  const int tx = threadIdx.x & 31, ty = threadIdx.x >> 5;  // 32 x 8
  for (int i = ty; i < 32; i += 8)
    tile[i][tx] = W[(size_t)(kb + i) * N + (nb + tx)];
  __syncthreads();
  for (int i = ty; i < 32; i += 8)
    Wt[(size_t)(nb + i) * K + (kb + tx)] = f2bf(tile[tx][i]);
}

// ---------------------------------------------------------------------------
// LayerNorm over C per (b,t) row; optionally capture row t==T-1 into lastOut
// ---------------------------------------------------------------------------
__global__ __launch_bounds__(256)
void k_layernorm(const float* __restrict__ X, const float* __restrict__ w,
                 const float* __restrict__ bb, float* __restrict__ Y,
                 float* __restrict__ lastOut) {
  const int row = blockIdx.x;                       // 0..BT-1
  const float* xr = X + (size_t)row * C_;
  __shared__ float s1[256], s2[256];
  float sum = 0.f, sq = 0.f;
  for (int c = threadIdx.x; c < C_; c += 256) {
    float v = xr[c]; sum += v; sq += v * v;
  }
  s1[threadIdx.x] = sum; s2[threadIdx.x] = sq;
  __syncthreads();
  for (int s = 128; s > 0; s >>= 1) {
    if (threadIdx.x < s) { s1[threadIdx.x] += s1[threadIdx.x + s];
                           s2[threadIdx.x] += s2[threadIdx.x + s]; }
    __syncthreads();
  }
  const float mean = s1[0] * (1.f / C_);
  const float var  = s2[0] * (1.f / C_) - mean * mean;
  const float inv  = rsqrtf(var + EPS_);
  const bool  last = ((row % T_) == (T_ - 1));
  const int   b    = row / T_;
  for (int c = threadIdx.x; c < C_; c += 256) {
    float v = (xr[c] - mean) * inv * w[c] + bb[c];
    Y[(size_t)row * C_ + c] = v;
    if (last && lastOut) lastOut[(size_t)b * C_ + c] = v;
  }
}

// ---------------------------------------------------------------------------
// Token-shift + 4-way mix -> bf16 (attention path)
// ---------------------------------------------------------------------------
__global__ __launch_bounds__(256)
void k_mix4(const float* __restrict__ x1, const float* __restrict__ shift,
            const float* __restrict__ mk, const float* __restrict__ mv,
            const float* __restrict__ mr, const float* __restrict__ mg,
            __bf16* __restrict__ xk, __bf16* __restrict__ xv,
            __bf16* __restrict__ xr, __bf16* __restrict__ xg) {
  size_t idx = (size_t)blockIdx.x * 256 + threadIdx.x;
  if (idx >= (size_t)BT_ * C_) return;
  const int c   = (int)(idx % C_);
  const size_t row = idx / C_;
  const int t = (int)(row % T_);
  const int b = (int)(row / T_);
  const float cur  = x1[idx];
  const float prev = (t == 0) ? shift[(size_t)b * C_ + c] : x1[idx - C_];
  const float k = mk[c], v = mv[c], r = mr[c], g = mg[c];
  xk[idx] = f2bf(cur * k + prev * (1.f - k));
  xv[idx] = f2bf(cur * v + prev * (1.f - v));
  xr[idx] = f2bf(cur * r + prev * (1.f - r));
  xg[idx] = f2bf(cur * g + prev * (1.f - g));
}

// Token-shift + 2-way mix -> bf16 (FFN path)
__global__ __launch_bounds__(256)
void k_mix2(const float* __restrict__ x2, const float* __restrict__ shift,
            const float* __restrict__ mk, const float* __restrict__ mr,
            __bf16* __restrict__ xck, __bf16* __restrict__ xcr) {
  size_t idx = (size_t)blockIdx.x * 256 + threadIdx.x;
  if (idx >= (size_t)BT_ * C_) return;
  const int c   = (int)(idx % C_);
  const size_t row = idx / C_;
  const int t = (int)(row % T_);
  const int b = (int)(row / T_);
  const float cur  = x2[idx];
  const float prev = (t == 0) ? shift[(size_t)b * C_ + c] : x2[idx - C_];
  const float k = mk[c], r = mr[c];
  xck[idx] = f2bf(cur * k + prev * (1.f - k));
  xcr[idx] = f2bf(cur * r + prev * (1.f - r));
}

// ---------------------------------------------------------------------------
// WMMA bf16 GEMM:  C[M x N] = act(A[M x K] @ Bt[N x K]^T) (+ resid)
// Block tile 128(M) x 256(N), 256 threads = 8 waves as 2(M) x 4(N);
// each wave owns 64x64 = 4x4 V_WMMA_F32_16X16X32_BF16 per 32-wide K step.
// Tiles staged HBM->LDS with GLOBAL_LOAD_ASYNC_TO_LDS_B128 (ASYNCcnt),
// double-buffered with one barrier per K step.
// ---------------------------------------------------------------------------
#define ACT_NONE    0
#define ACT_SILU    1
#define ACT_RELUSQ  2
#define ACT_SIGMOID 3

template <int ACT>
__global__ __launch_bounds__(256)
void k_gemm_bf16(const __bf16* __restrict__ A, const __bf16* __restrict__ Bt,
                 float* __restrict__ Cf, __bf16* __restrict__ Cbf,
                 const float* __restrict__ resid, int N, int K) {
  __shared__ __align__(16) __bf16 As[2][128][32];
  __shared__ __align__(16) __bf16 Bs[2][256][32];

  const int tid  = threadIdx.x;
  const int lane = tid & 31;
  const int wid  = tid >> 5;
  const int wm   = (wid & 1) * 64;      // wave row offset inside tile
  const int wn   = (wid >> 1) * 64;     // wave col offset inside tile
  const int rowBase = blockIdx.y * 128;
  const int colBase = blockIdx.x * 256;

  f32x8 acc[4][4];
#pragma unroll
  for (int mi = 0; mi < 4; ++mi)
#pragma unroll
    for (int ni = 0; ni < 4; ++ni)
      acc[mi][ni] = (f32x8){0.f,0.f,0.f,0.f,0.f,0.f,0.f,0.f};

  const int rsel = lane & 15;
  const int kA   = (lane & 16) ? 8  : 0;   // 16-bit A 16x32 fragment layout
  const int kB   = (lane & 16) ? 8  : 0;   // 16-bit B 32x16 layout (K-major LDS)

  // Async-issue one K-step tile pair into buffer `buf`.
  // A tile: 128x32 bf16 = 512 16B-chunks; B tile: 256x32 = 1024 chunks.
  auto issue = [&](int buf, int k0) {
#pragma unroll
    for (int i = 0; i < 2; ++i) {
      const int idx = tid + i * 256;
      const int r = idx >> 2, q = (idx & 3) * 8;
      async_ld16((uint32_t)(uintptr_t)&As[buf][r][q],
                 &A[(size_t)(rowBase + r) * K + k0 + q]);
    }
#pragma unroll
    for (int i = 0; i < 4; ++i) {
      const int idx = tid + i * 256;
      const int r = idx >> 2, q = (idx & 3) * 8;
      async_ld16((uint32_t)(uintptr_t)&Bs[buf][r][q],
                 &Bt[(size_t)(colBase + r) * K + k0 + q]);
    }
  };

  issue(0, 0);
  const int steps = K / 32;
  for (int s = 0; s < steps; ++s) {
    const int buf = s & 1;
    wait_async0();          // our tile chunks have landed in LDS
    __syncthreads();        // everyone's chunks landed; prev reads finished
    if (s + 1 < steps) issue(buf ^ 1, (s + 1) * 32);

    bf16x16 af[4];
#pragma unroll
    for (int mi = 0; mi < 4; ++mi) {
      union { bf16x16 v; uint4 u[2]; } t;
      t.u[0] = *(const uint4*)&As[buf][wm + mi * 16 + rsel][kA];
      t.u[1] = *(const uint4*)&As[buf][wm + mi * 16 + rsel][kA + 16];
      af[mi] = t.v;
    }
#pragma unroll
    for (int ni = 0; ni < 4; ++ni) {
      union { bf16x16 v; uint4 u[2]; } t;
      t.u[0] = *(const uint4*)&Bs[buf][wn + ni * 16 + rsel][kB];
      t.u[1] = *(const uint4*)&Bs[buf][wn + ni * 16 + rsel][kB + 16];
      const bf16x16 bfr = t.v;
#pragma unroll
      for (int mi = 0; mi < 4; ++mi)
        acc[mi][ni] = __builtin_amdgcn_wmma_f32_16x16x32_bf16(
            false, af[mi], false, bfr, (short)0, acc[mi][ni], false, false);
    }
  }

  // Epilogue: C/D 16x16 f32 layout -> row = e + 8*(lane>=16), col = lane&15
  const int cEl  = lane & 15;
  const int rAdd = (lane & 16) ? 8 : 0;
#pragma unroll
  for (int mi = 0; mi < 4; ++mi)
#pragma unroll
    for (int ni = 0; ni < 4; ++ni)
#pragma unroll
      for (int e = 0; e < 8; ++e) {
        const int row = rowBase + wm + mi * 16 + rAdd + e;
        const int col = colBase + wn + ni * 16 + cEl;
        float v = acc[mi][ni][e];
        if (ACT == ACT_SILU)    v = v / (1.f + __expf(-v));
        if (ACT == ACT_RELUSQ)  { float t = fmaxf(v, 0.f); v = t * t; }
        if (ACT == ACT_SIGMOID) v = 1.f / (1.f + __expf(-v));
        const size_t o = (size_t)row * N + col;
        if (resid) v += resid[o];
        if (Cf)  Cf[o]  = v;
        if (Cbf) Cbf[o] = f2bf(v);
      }
}

// ---------------------------------------------------------------------------
// WKV linear-attention scan. One block per (b,h); 64 threads; thread j owns
// state column S[:, j] (64 f32 in registers). Sequential over T.
// ---------------------------------------------------------------------------
__global__ __launch_bounds__(64)
void k_wkv(const float* __restrict__ r, const float* __restrict__ k,
           const float* __restrict__ v, const float* __restrict__ Sin_all,
           const float* __restrict__ time_decay, const float* __restrict__ time_faaaa,
           float* __restrict__ att, float* __restrict__ Sout_all) {
  const int bh = blockIdx.x;            // b*H + h
  const int b  = bh / H_, h = bh % H_;
  const int j  = threadIdx.x;

  __shared__ float ws[N_], us[N_], rs[N_], ks[N_], vs[N_];
  ws[j] = __expf(-__expf(time_decay[h * N_ + j]));
  us[j] = time_faaaa[h * N_ + j];

  float S[N_];
  const float* Sin = Sin_all + (size_t)bh * N_ * N_;
#pragma unroll
  for (int i = 0; i < N_; ++i) S[i] = Sin[i * N_ + j];
  __syncthreads();

  for (int t = 0; t < T_; ++t) {
    const size_t base = ((size_t)(b * T_ + t)) * C_ + h * N_;
    rs[j] = r[base + j]; ks[j] = k[base + j]; vs[j] = v[base + j];
    __syncthreads();
    const float vj = vs[j];
    float y = 0.f;
#pragma unroll
    for (int i = 0; i < N_; ++i) {
      const float kv = ks[i] * vj;
      y += rs[i] * (S[i] + us[i] * kv);
      S[i] = S[i] * ws[i] + kv;
    }
    att[base + j] = y;
    __syncthreads();
  }
  float* Sout = Sout_all + (size_t)bh * N_ * N_;
#pragma unroll
  for (int i = 0; i < N_; ++i) Sout[i * N_ + j] = S[i];
}

// ---------------------------------------------------------------------------
// GroupNorm per (b,t,h) over N=64 of att/8, then gate with silu(g) -> bf16
// ---------------------------------------------------------------------------
__global__ __launch_bounds__(64)
void k_gngate(const float* __restrict__ att, const float* __restrict__ g,
              const float* __restrict__ gnw, const float* __restrict__ gnb,
              __bf16* __restrict__ gated) {
  const int bth = blockIdx.x;           // (b*T + t)*H + h
  const int h   = bth % H_;
  const int n   = threadIdx.x;
  const size_t base = (size_t)(bth / H_) * C_ + h * N_;

  const float z = att[base + n] * (1.f / 8.f);
  __shared__ float s1[N_], s2[N_];
  s1[n] = z; s2[n] = z * z;
  __syncthreads();
  for (int s = 32; s > 0; s >>= 1) {
    if (n < s) { s1[n] += s1[n + s]; s2[n] += s2[n + s]; }
    __syncthreads();
  }
  const float mean = s1[0] * (1.f / N_);
  const float var  = s2[0] * (1.f / N_) - mean * mean;
  const float inv  = rsqrtf(var + EPS_);
  const float zz   = (z - mean) * inv * gnw[h * N_ + n] + gnb[h * N_ + n];
  gated[base + n]  = f2bf(zz * g[base + n]);
}

// out = xres + sr * kv
__global__ __launch_bounds__(256)
void k_combine(const float* __restrict__ xres, const float* __restrict__ sr,
               const float* __restrict__ kv, float* __restrict__ out) {
  size_t idx = (size_t)blockIdx.x * 256 + threadIdx.x;
  if (idx >= (size_t)BT_ * C_) return;
  out[idx] = xres[idx] + sr[idx] * kv[idx];
}

// ---------------------------------------------------------------------------
extern "C" void kernel_launch(void* const* d_in, const int* in_sizes, int n_in,
                              void* d_out, int out_size, void* d_ws, size_t ws_size,
                              hipStream_t stream) {
  const float* x         = (const float*)d_in[0];
  const float* att_shift = (const float*)d_in[1];
  const float* wkv_state = (const float*)d_in[2];
  const float* ffn_shift = (const float*)d_in[3];
  const float* ln1_w = (const float*)d_in[4];
  const float* ln1_b = (const float*)d_in[5];
  const float* ln2_w = (const float*)d_in[6];
  const float* ln2_b = (const float*)d_in[7];
  const float* mix_k = (const float*)d_in[8];
  const float* mix_v = (const float*)d_in[9];
  const float* mix_r = (const float*)d_in[10];
  const float* mix_g = (const float*)d_in[11];
  const float* Wr = (const float*)d_in[12];
  const float* Wk = (const float*)d_in[13];
  const float* Wv = (const float*)d_in[14];
  const float* Wg = (const float*)d_in[15];
  const float* Wo = (const float*)d_in[16];
  const float* time_decay = (const float*)d_in[17];
  const float* time_faaaa = (const float*)d_in[18];
  const float* gn_w = (const float*)d_in[19];
  const float* gn_b = (const float*)d_in[20];
  const float* cmix_k = (const float*)d_in[21];
  const float* cmix_r = (const float*)d_in[22];
  const float* Wck = (const float*)d_in[23];
  const float* Wcr = (const float*)d_in[24];
  const float* Wcv = (const float*)d_in[25];

  // Output layout: x (BT*C) | x1_last (B*C) | new_wkv (B*H*N*N) | x2_last (B*C)
  float* out_x   = (float*)d_out;
  float* out_x1l = out_x   + (size_t)BT_ * C_;
  float* out_wkv = out_x1l + (size_t)B_ * C_;
  float* out_x2l = out_wkv + (size_t)B_ * H_ * N_ * N_;

  // Workspace bump allocator (256B aligned)
  char*  ws  = (char*)d_ws;
  size_t off = 0;
  auto alloc = [&](size_t bytes) -> void* {
    off = (off + 255) & ~(size_t)255;
    void* p = ws + off;
    off += bytes;
    return p;
  };
  const size_t BTC = (size_t)BT_ * C_;
  const size_t CC  = (size_t)C_ * C_;
  const size_t CF  = (size_t)C_ * F_;

  __bf16* WrT  = (__bf16*)alloc(CC * 2);
  __bf16* WkT  = (__bf16*)alloc(CC * 2);
  __bf16* WvT  = (__bf16*)alloc(CC * 2);
  __bf16* WgT  = (__bf16*)alloc(CC * 2);
  __bf16* WoT  = (__bf16*)alloc(CC * 2);
  __bf16* WckT = (__bf16*)alloc(CF * 2);   // (F x C)
  __bf16* WcrT = (__bf16*)alloc(CC * 2);
  __bf16* WcvT = (__bf16*)alloc(CF * 2);   // (C x F)

  float*  x1   = (float*)alloc(BTC * 4);
  __bf16* xk   = (__bf16*)alloc(BTC * 2);
  __bf16* xv   = (__bf16*)alloc(BTC * 2);
  __bf16* xr   = (__bf16*)alloc(BTC * 2);
  __bf16* xg   = (__bf16*)alloc(BTC * 2);
  float*  rb   = (float*)alloc(BTC * 4);
  float*  kb   = (float*)alloc(BTC * 4);
  float*  vb   = (float*)alloc(BTC * 4);
  float*  gb   = (float*)alloc(BTC * 4);
  float*  attb = (float*)alloc(BTC * 4);
  __bf16* gated= (__bf16*)alloc(BTC * 2);
  float*  xres = (float*)alloc(BTC * 4);
  float*  x2   = (float*)alloc(BTC * 4);
  __bf16* xck  = (__bf16*)alloc(BTC * 2);
  __bf16* xcr  = (__bf16*)alloc(BTC * 2);
  __bf16* kkb  = (__bf16*)alloc((size_t)BT_ * F_ * 2);
  float*  kvb  = (float*)alloc(BTC * 4);
  float*  srb  = (float*)alloc(BTC * 4);

  // --- 1. weights -> bf16, K-major (transposed) ---
  auto launchT = [&](const float* W, __bf16* Wt, int K, int N) {
    dim3 g(N / 32, K / 32);
    k_transpose_bf16<<<g, 256, 0, stream>>>(W, Wt, K, N);
  };
  launchT(Wr,  WrT,  C_, C_);
  launchT(Wk,  WkT,  C_, C_);
  launchT(Wv,  WvT,  C_, C_);
  launchT(Wg,  WgT,  C_, C_);
  launchT(Wo,  WoT,  C_, C_);
  launchT(Wck, WckT, C_, F_);
  launchT(Wcr, WcrT, C_, C_);
  launchT(Wcv, WcvT, F_, C_);

  const int elemBlocks = (int)((BTC + 255) / 256);

  // --- 2. LN1 + mixes ---
  k_layernorm<<<BT_, 256, 0, stream>>>(x, ln1_w, ln1_b, x1, out_x1l);
  k_mix4<<<elemBlocks, 256, 0, stream>>>(x1, att_shift, mix_k, mix_v, mix_r,
                                         mix_g, xk, xv, xr, xg);

  // --- 3. r/k/v/g projections (WMMA) ---
  dim3 gC(C_ / 256, BT_ / 128);   // 256-wide N tiles x 128-wide M tiles
  k_gemm_bf16<ACT_NONE><<<gC, 256, 0, stream>>>(xr, WrT, rb, (__bf16*)nullptr,
                                                (const float*)nullptr, C_, C_);
  k_gemm_bf16<ACT_NONE><<<gC, 256, 0, stream>>>(xk, WkT, kb, (__bf16*)nullptr,
                                                (const float*)nullptr, C_, C_);
  k_gemm_bf16<ACT_NONE><<<gC, 256, 0, stream>>>(xv, WvT, vb, (__bf16*)nullptr,
                                                (const float*)nullptr, C_, C_);
  k_gemm_bf16<ACT_SILU><<<gC, 256, 0, stream>>>(xg, WgT, gb, (__bf16*)nullptr,
                                                (const float*)nullptr, C_, C_);

  // --- 4. WKV scan (writes final state straight into d_out) ---
  k_wkv<<<B_ * H_, 64, 0, stream>>>(rb, kb, vb, wkv_state, time_decay,
                                    time_faaaa, attb, out_wkv);

  // --- 5. GroupNorm + gate, output projection with fused residual ---
  k_gngate<<<BT_ * H_, 64, 0, stream>>>(attb, gb, gn_w, gn_b, gated);
  k_gemm_bf16<ACT_NONE><<<gC, 256, 0, stream>>>(gated, WoT, xres,
                                                (__bf16*)nullptr, x, C_, C_);

  // --- 6. LN2 + FFN mixes ---
  k_layernorm<<<BT_, 256, 0, stream>>>(xres, ln2_w, ln2_b, x2, out_x2l);
  k_mix2<<<elemBlocks, 256, 0, stream>>>(x2, ffn_shift, cmix_k, cmix_r, xck, xcr);

  // --- 7. FFN: kk = relu(xck@Wck)^2 (bf16), kv = kk@Wcv, sr = sigmoid(xcr@Wcr)
  dim3 gF(F_ / 256, BT_ / 128);
  k_gemm_bf16<ACT_RELUSQ><<<gF, 256, 0, stream>>>(xck, WckT, (float*)nullptr,
                                                  kkb, (const float*)nullptr,
                                                  F_, C_);
  k_gemm_bf16<ACT_NONE><<<gC, 256, 0, stream>>>(kkb, WcvT, kvb, (__bf16*)nullptr,
                                                (const float*)nullptr, C_, F_);
  k_gemm_bf16<ACT_SIGMOID><<<gC, 256, 0, stream>>>(xcr, WcrT, srb,
                                                   (__bf16*)nullptr,
                                                   (const float*)nullptr, C_, C_);

  // --- 8. final combine into d_out ---
  k_combine<<<elemBlocks, 256, 0, stream>>>(xres, srb, kvb, out_x);
}
